// SparseAttention_40948218200576
// MI455X (gfx1250) — matrix operand order repaired and checked by reading, not compiled, not run on hardware
//
#include <hip/hip_runtime.h>

// ---------------------------------------------------------------------------
// Types for CDNA5 WMMA (wave32): v_wmma_f32_16x16x32_bf16
// ---------------------------------------------------------------------------
typedef __attribute__((ext_vector_type(16))) __bf16 v16bf;
typedef __attribute__((ext_vector_type(8)))  __bf16 v8bf;
typedef __attribute__((ext_vector_type(8)))  float  v8f;

union BF16x16 { v16bf v; v8bf h[2]; };

constexpr int Bn = 2;
constexpr int Tn = 2048;
constexpr int Cn = 1024;
constexpr int Hn = 8;
constexpr int Dn = 128;
constexpr int Mrows = Bn * Tn;                 // 4096 token rows
constexpr float kScale = 0.0883883476483184f;  // 1/sqrt(128)

__device__ __forceinline__ int imin(int a, int b) { return a < b ? a : b; }
__device__ __forceinline__ int imax(int a, int b) { return a > b ? a : b; }

// 32-bit LDS byte address from a generic pointer to __shared__ memory.
__device__ __forceinline__ unsigned lds_addr32(const void* p) {
  return (unsigned)(size_t)(__attribute__((address_space(3))) const void*)p;
}

// CDNA5 async copy: 16 bytes global -> LDS, tracked by ASYNCcnt.
__device__ __forceinline__ void async_copy_b128(unsigned lds_byte_addr,
                                                const void* gptr) {
  asm volatile("global_load_async_to_lds_b128 %0, %1, off"
               :: "v"(lds_byte_addr), "v"(gptr)
               : "memory");
}
__device__ __forceinline__ void wait_async0() {
  asm volatile("s_wait_asynccnt 0" ::: "memory");
}

// ---------------------------------------------------------------------------
// fp32 -> bf16 conversion (plain)
// ---------------------------------------------------------------------------
__global__ void f32_to_bf16_kernel(const float* __restrict__ in,
                                   __bf16* __restrict__ out, int n) {
  int i = blockIdx.x * blockDim.x + threadIdx.x;
  if (i < n) out[i] = (__bf16)in[i];
}

// fp32 [rows][cols] -> bf16 transposed [cols][rows]; coalesced reads.
__global__ void f32_to_bf16_T_kernel(const float* __restrict__ in,
                                     __bf16* __restrict__ out,
                                     int rows, int cols) {
  int i = blockIdx.x * blockDim.x + threadIdx.x;
  if (i < rows * cols) {
    int r = i / cols, c = i % cols;
    out[(size_t)c * rows + r] = (__bf16)in[i];
  }
}

// ---------------------------------------------------------------------------
// 64x128 block-tile bf16 GEMM core, B supplied PRE-TRANSPOSED (Bt[N][K]),
// K = Cn (compile-time). 128 threads = 4 waves; each wave owns a 32x64
// quadrant = 2x4 WMMA tiles (8 v_wmma per 32-wide K step). Staging uses
// CDNA5 async global->LDS copies with double-buffered LDS in an explicit
// two-phase steady state (K stepped by 64): wait_async -> barrier ->
// issue next buffer's asyncs -> fragment loads + WMMAs.
// ---------------------------------------------------------------------------
__device__ __forceinline__ void gemm_stage_async(
    const __bf16* __restrict__ A, const __bf16* __restrict__ Bt,
    int m0, int n0, int kb, __bf16* sA, __bf16* sBt) {
  const int tid = threadIdx.x;
  { // A tile 64x32: each thread one 32B chunk
    const int row = tid >> 1, half = tid & 1;
    const __bf16* src = A + (size_t)(m0 + row) * Cn + kb + 16 * half;
    const unsigned dst = lds_addr32(sA + row * 32 + 16 * half);
    async_copy_b128(dst, src);
    async_copy_b128(dst + 16, src + 8);
  }
  { // Bt tile 128x32 (K-contiguous rows): each thread one 64B row chunk
    const __bf16* src = Bt + (size_t)(n0 + tid) * Cn + kb;
    const unsigned dst = lds_addr32(sBt + tid * 32);
#pragma unroll
    for (int e = 0; e < 4; ++e) async_copy_b128(dst + 16 * e, src + 8 * e);
  }
}

__device__ __forceinline__ void gemm_compute_step(
    const __bf16* cA, const __bf16* cB,
    int wm, int wn, int lm, int lh, v8f acc[2][4]) {
  BF16x16 af[2], bf[4];
#pragma unroll
  for (int i = 0; i < 2; ++i) { // A frag: K = 8*lh+e (e<8), 16+8*lh+(e-8)
    const __bf16* ap = cA + (wm + i * 16 + lm) * 32;
    af[i].h[0] = *(const v8bf*)(ap + 8 * lh);
    af[i].h[1] = *(const v8bf*)(ap + 16 + 8 * lh);
  }
#pragma unroll
  for (int j = 0; j < 4; ++j) { // B frag: K = 8*lh+e (e<8), 16+8*lh+(e-8)
    const __bf16* bp = cB + (wn + j * 16 + lm) * 32;
    bf[j].h[0] = *(const v8bf*)(bp + 8 * lh);
    bf[j].h[1] = *(const v8bf*)(bp + 16 + 8 * lh);
  }
#pragma unroll
  for (int i = 0; i < 2; ++i)
#pragma unroll
    for (int j = 0; j < 4; ++j)
      acc[i][j] = __builtin_amdgcn_wmma_f32_16x16x32_bf16(
          false, af[i].v, false, bf[j].v, (short)0, acc[i][j], false, false);
}

__device__ __forceinline__ void gemm_core_64x128(
    const __bf16* __restrict__ A, const __bf16* __restrict__ Bt,
    int m0, int n0, __bf16* sA, __bf16* sBt, v8f acc[2][4]) {
  const int lane = threadIdx.x & 31;
  const int w    = threadIdx.x >> 5;
  const int wm   = (w >> 1) * 32;
  const int wn   = (w & 1) * 64;
  const int lm   = lane & 15;
  const int lh   = lane >> 4;

  __bf16* A0 = sA;             __bf16* B0 = sBt;
  __bf16* A1 = sA + 64 * 32;   __bf16* B1 = sBt + 128 * 32;

  gemm_stage_async(A, Bt, m0, n0, 0, A0, B0); // prologue: buffer 0

  for (int kb = 0; kb < Cn; kb += 64) {
    // phase 0: consume buffer 0, refill buffer 1 (always valid: kb+32 < Cn)
    wait_async0();
    __syncthreads();
    gemm_stage_async(A, Bt, m0, n0, kb + 32, A1, B1);
    gemm_compute_step(A0, B0, wm, wn, lm, lh, acc);

    // phase 1: consume buffer 1, refill buffer 0 (skip on final step)
    wait_async0();
    __syncthreads();
    if (kb + 64 < Cn)
      gemm_stage_async(A, Bt, m0, n0, kb + 64, A0, B0);
    gemm_compute_step(A1, B1, wm, wn, lm, lh, acc);
  }
}

// ---------------------------------------------------------------------------
// QKV GEMM: [4096,1024] x Wqkv^T[3072][1024] + bias, scatter epilogue into
//   Q [B,H,T,D], K [B,H,T,D], V^T [B,H,D,T]  (all bf16)
// ---------------------------------------------------------------------------
__global__ void __launch_bounds__(128) gemm_qkv_kernel(
    const __bf16* __restrict__ A, const __bf16* __restrict__ Bt,
    const float* __restrict__ bias,
    __bf16* __restrict__ qws, __bf16* __restrict__ kws,
    __bf16* __restrict__ vtws) {
  __shared__ __bf16 sA[2 * 64 * 32];
  __shared__ __bf16 sBt[2 * 128 * 32];
  const int m0 = blockIdx.y * 64, n0 = blockIdx.x * 128;
  v8f acc[2][4] = {};
  gemm_core_64x128(A, Bt, m0, n0, sA, sBt, acc);

  const int lane = threadIdx.x & 31;
  const int w    = threadIdx.x >> 5;
  const int wm   = (w >> 1) * 32, wn = (w & 1) * 64;
  const int lm   = lane & 15, lh = lane >> 4;
#pragma unroll
  for (int i = 0; i < 2; ++i)
#pragma unroll
    for (int j = 0; j < 4; ++j)
#pragma unroll
      for (int v = 0; v < 8; ++v) {
        const int row = m0 + wm + i * 16 + v + 8 * lh; // C-layout row
        const int col = n0 + wn + j * 16 + lm;         // C-layout col
        const float val = acc[i][j][v] + bias[col];
        const int which = col >> 10, jj = col & (Cn - 1);
        const int head = jj >> 7, dim = jj & (Dn - 1);
        const int bb = row >> 11, t = row & (Tn - 1);
        const size_t bh = (size_t)bb * Hn + head;
        if (which == 0)      qws[(bh * Tn + t) * Dn + dim]  = (__bf16)val;
        else if (which == 1) kws[(bh * Tn + t) * Dn + dim]  = (__bf16)val;
        else                 vtws[(bh * Dn + dim) * Tn + t] = (__bf16)val;
      }
}

// ---------------------------------------------------------------------------
// Output projection GEMM: [4096,1024] x Wproj^T[1024][1024] + bias -> fp32
// ---------------------------------------------------------------------------
__global__ void __launch_bounds__(128) gemm_proj_kernel(
    const __bf16* __restrict__ A, const __bf16* __restrict__ Bt,
    const float* __restrict__ bias, float* __restrict__ out) {
  __shared__ __bf16 sA[2 * 64 * 32];
  __shared__ __bf16 sBt[2 * 128 * 32];
  const int m0 = blockIdx.y * 64, n0 = blockIdx.x * 128;
  v8f acc[2][4] = {};
  gemm_core_64x128(A, Bt, m0, n0, sA, sBt, acc);

  const int lane = threadIdx.x & 31;
  const int w    = threadIdx.x >> 5;
  const int wm   = (w >> 1) * 32, wn = (w & 1) * 64;
  const int lm   = lane & 15, lh = lane >> 4;
#pragma unroll
  for (int i = 0; i < 2; ++i)
#pragma unroll
    for (int j = 0; j < 4; ++j)
#pragma unroll
      for (int v = 0; v < 8; ++v) {
        const int row = m0 + wm + i * 16 + v + 8 * lh;
        const int col = n0 + wn + j * 16 + lm;
        out[(size_t)row * Cn + col] = acc[i][j][v] + bias[col];
      }
}

// ---------------------------------------------------------------------------
// Sliding-window causal attention, flash-style online softmax.
// One wave per 16-query tile per (b,h). 5 key blocks of 32 cover
// [q0-128, q0+31]; invalid keys masked (exp -> 0). P re-laid-out through a
// per-wave LDS slice from C-format to A-fragment format.
// ---------------------------------------------------------------------------
__global__ void __launch_bounds__(128) attn_kernel(
    const __bf16* __restrict__ q, const __bf16* __restrict__ k,
    const __bf16* __restrict__ vt, const float* __restrict__ gate,
    __bf16* __restrict__ attw) {
  __shared__ __bf16 ldsP[4][16 * 32]; // per-wave P staging (16 rows x 32 keys)
  const int w    = threadIdx.x >> 5;
  const int lane = threadIdx.x & 31;
  const int lm   = lane & 15, lh = lane >> 4;

  const int tile = blockIdx.x * 4 + w;
  const int q0   = (tile & (Tn / 16 - 1)) * 16;
  const int bh   = tile >> 7;               // Tn/16 == 128 tiles per (b,h)
  const int bb   = bh >> 3, hh = bh & 7;

  const __bf16* qp  = q  + (size_t)bh * Tn * Dn;
  const __bf16* kp  = k  + (size_t)bh * Tn * Dn;
  const __bf16* vtp = vt + (size_t)bh * Dn * Tn;

  // Q fragments for all 4 K-chunks of D=128, cached in registers.
  BF16x16 qf[4];
#pragma unroll
  for (int d4 = 0; d4 < 4; ++d4) {
    const __bf16* base = qp + (size_t)(q0 + lm) * Dn + 32 * d4;
    qf[d4].h[0] = *(const v8bf*)(base + 8 * lh);
    qf[d4].h[1] = *(const v8bf*)(base + 16 + 8 * lh);
  }

  v8f O[8] = {};
  float mrow[8], lrow[8];
#pragma unroll
  for (int v = 0; v < 8; ++v) { mrow[v] = -1e30f; lrow[v] = 0.f; }

  for (int blk = 0; blk < 5; ++blk) {
    const int kb = q0 - 128 + 32 * blk;
    if (kb + 32 <= 0) continue; // block fully left of sequence

    // ---- S = Q K^T for two 16-key tiles ----
    v8f s0 = {}, s1 = {};
    const int key0 = kb + lm, key1 = kb + 16 + lm;
    const int k0c = imin(imax(key0, 0), Tn - 1);
    const int k1c = imin(imax(key1, 0), Tn - 1);
#pragma unroll
    for (int d4 = 0; d4 < 4; ++d4) {
      BF16x16 kf0, kf1;
      const __bf16* p0 = kp + (size_t)k0c * Dn + 32 * d4;
      const __bf16* p1 = kp + (size_t)k1c * Dn + 32 * d4;
      kf0.h[0] = *(const v8bf*)(p0 + 8 * lh); kf0.h[1] = *(const v8bf*)(p0 + 16 + 8 * lh);
      kf1.h[0] = *(const v8bf*)(p1 + 8 * lh); kf1.h[1] = *(const v8bf*)(p1 + 16 + 8 * lh);
      s0 = __builtin_amdgcn_wmma_f32_16x16x32_bf16(false, qf[d4].v, false, kf0.v,
                                                   (short)0, s0, false, false);
      s1 = __builtin_amdgcn_wmma_f32_16x16x32_bf16(false, qf[d4].v, false, kf1.v,
                                                   (short)0, s1, false, false);
    }

    // ---- online softmax update (C-layout: row = v + 8*lh, col = lm) ----
    float fscale[8];
#pragma unroll
    for (int v = 0; v < 8; ++v) {
      const int row = q0 + v + 8 * lh;
      const bool valid0 = (key0 >= 0) && (key0 <= row) && (row - key0 <= 128);
      const bool valid1 = (key1 >= 0) && (key1 <= row) && (row - key1 <= 128);
      float a0 = valid0 ? s0[v] * kScale : -1e30f;
      float a1 = valid1 ? s1[v] * kScale : -1e30f;
      float bm = fmaxf(a0, a1);
      bm = fmaxf(bm, __shfl_xor(bm, 1));
      bm = fmaxf(bm, __shfl_xor(bm, 2));
      bm = fmaxf(bm, __shfl_xor(bm, 4));
      bm = fmaxf(bm, __shfl_xor(bm, 8));
      const float mn = fmaxf(mrow[v], bm);
      const float f  = __expf(mrow[v] - mn);
      const float p0 = valid0 ? __expf(a0 - mn) : 0.f;
      const float p1 = valid1 ? __expf(a1 - mn) : 0.f;
      float ps = p0 + p1;
      ps += __shfl_xor(ps, 1);
      ps += __shfl_xor(ps, 2);
      ps += __shfl_xor(ps, 4);
      ps += __shfl_xor(ps, 8);
      lrow[v] = lrow[v] * f + ps;
      mrow[v] = mn;
      fscale[v] = f;
      ldsP[w][(v + 8 * lh) * 32 + lm]      = (__bf16)p0;
      ldsP[w][(v + 8 * lh) * 32 + 16 + lm] = (__bf16)p1;
    }
#pragma unroll
    for (int c = 0; c < 8; ++c)
#pragma unroll
      for (int v = 0; v < 8; ++v) O[c][v] *= fscale[v];

    asm volatile("s_wait_dscnt 0" ::: "memory"); // LDS write->read, same wave

    // ---- P fragment (A-layout) from LDS ----
    BF16x16 pf;
    pf.h[0] = *(const v8bf*)(&ldsP[w][lm * 32 + 16 * lh]);
    pf.h[1] = *(const v8bf*)(&ldsP[w][lm * 32 + 16 * lh + 8]);

    // ---- O += P V : V^T layout makes B-fragments contiguous ----
    const int ks = imin(imax(kb + 16 * lh, 0), Tn - 16); // 16-aligned segment
#pragma unroll
    for (int c = 0; c < 8; ++c) {
      const __bf16* vp = vtp + (size_t)(16 * c + lm) * Tn + ks;
      BF16x16 vf;
      vf.h[0] = *(const v8bf*)(vp);
      vf.h[1] = *(const v8bf*)(vp + 8);
      O[c] = __builtin_amdgcn_wmma_f32_16x16x32_bf16(false, pf.v, false, vf.v,
                                                     (short)0, O[c], false, false);
    }
  }

  // ---- epilogue: normalize, gate, store bf16 [B*T, C] head-concat ----
#pragma unroll
  for (int c = 0; c < 8; ++c) {
    const int ch = hh * Dn + 16 * c + lm;
    const float g = gate[ch];
#pragma unroll
    for (int v = 0; v < 8; ++v) {
      const int t = q0 + v + 8 * lh;
      attw[((size_t)bb * Tn + t) * Cn + ch] = (__bf16)(O[c][v] / lrow[v] * g);
    }
  }
}

// ---------------------------------------------------------------------------
// Host launcher
// ---------------------------------------------------------------------------
extern "C" void kernel_launch(void* const* d_in, const int* in_sizes, int n_in,
                              void* d_out, int out_size, void* d_ws, size_t ws_size,
                              hipStream_t stream) {
  (void)in_sizes; (void)n_in; (void)out_size; (void)ws_size;
  const float* x     = (const float*)d_in[0];
  const float* Wqkv  = (const float*)d_in[1];
  const float* bqkv  = (const float*)d_in[2];
  const float* Wproj = (const float*)d_in[3];
  const float* bproj = (const float*)d_in[4];
  const float* gate  = (const float*)d_in[5];
  float* out = (float*)d_out;

  char* ws = (char*)d_ws;
  size_t off = 0;
  __bf16* xbf    = (__bf16*)(ws + off); off += (size_t)Mrows * Cn * 2;        // 8 MB
  __bf16* wqkvT  = (__bf16*)(ws + off); off += (size_t)Cn * 3 * Cn * 2;       // 6 MB
  __bf16* wprjT  = (__bf16*)(ws + off); off += (size_t)Cn * Cn * 2;           // 2 MB
  __bf16* qws    = (__bf16*)(ws + off); off += (size_t)Bn * Hn * Tn * Dn * 2; // 8 MB
  __bf16* kws    = (__bf16*)(ws + off); off += (size_t)Bn * Hn * Tn * Dn * 2; // 8 MB
  __bf16* vtws   = (__bf16*)(ws + off); off += (size_t)Bn * Hn * Tn * Dn * 2; // 8 MB
  __bf16* attw   = (__bf16*)(ws + off); off += (size_t)Mrows * Cn * 2;        // 8 MB

  const int nx = Mrows * Cn, nwq = Cn * 3 * Cn, nwp = Cn * Cn;
  f32_to_bf16_kernel<<<(nx + 255) / 256, 256, 0, stream>>>(x, xbf, nx);
  // Weights converted AND transposed: Bt[N][K] so B-fragments are contiguous.
  f32_to_bf16_T_kernel<<<(nwq + 255) / 256, 256, 0, stream>>>(Wqkv, wqkvT, Cn, 3 * Cn);
  f32_to_bf16_T_kernel<<<(nwp + 255) / 256, 256, 0, stream>>>(Wproj, wprjT, Cn, Cn);

  gemm_qkv_kernel<<<dim3(3 * Cn / 128, Mrows / 64), 128, 0, stream>>>(
      xbf, wqkvT, bqkv, qws, kws, vtws);

  const int ntiles = Bn * Hn * (Tn / 16);            // 2048 wave-tiles
  attn_kernel<<<ntiles / 4, 128, 0, stream>>>(qws, kws, vtws, gate, attw);

  gemm_proj_kernel<<<dim3(Cn / 128, Mrows / 64), 128, 0, stream>>>(
      attw, wprjT, bproj, out);
}
